// MambaBlock_57277683859549
// MI455X (gfx1250) — compile-verified, matrix operand order checked by reading, not run on hardware
//
#include <hip/hip_runtime.h>
#include <hip/hip_bf16.h>
#include <math.h>

// ---------------------------------------------------------------------------
// Mamba block + FFN for gfx1250 (MI455X).  GEMMs: bf16 WMMA (16x16x32),
// 64x128 block tiles, double-buffered LDS staging via the Tensor Data Mover
// (tensor_load_to_lds / TENSORcnt) so DMA overlaps the WMMA compute phase.
// ---------------------------------------------------------------------------

typedef __bf16 v16bf __attribute__((ext_vector_type(16)));
typedef float  v8f   __attribute__((ext_vector_type(8)));
typedef unsigned int u32x4 __attribute__((ext_vector_type(4)));
typedef int          i32x8 __attribute__((ext_vector_type(8)));
typedef int          i32x4 __attribute__((ext_vector_type(4)));

#define MROWS 2048      // B*L
#define EMBED 768
#define DINNER 1536
#define DSTATE 16
#define DTRANK 48
#define FFDIM 3072
#define SEQL 1024
#define NBATCH 2

#define BM 64
#define BN 128
#define BK 32

enum { EPI_NONE = 0, EPI_BIAS_SOFTPLUS = 1, EPI_RESID = 2,
       EPI_BIAS_GELU = 3, EPI_BIAS_RESID = 4 };

#if defined(__has_builtin)
#  if __has_builtin(__builtin_amdgcn_tensor_load_to_lds)
#    define HAVE_TDM 1
#  endif
#endif
#ifndef HAVE_TDM
#  define HAVE_TDM 0
#endif

// -------------------------------------------------------------------- cvt ---
__global__ __launch_bounds__(256)
void cvt_pad_bf16(const float* __restrict__ in, __bf16* __restrict__ out,
                  int N, int K, int ld, int Kp, long long total)
{
    long long i = (long long)blockIdx.x * 256 + threadIdx.x;
    if (i >= total) return;
    int n = (int)(i / Kp);
    int k = (int)(i % Kp);
    float v = (n < N && k < K) ? in[(size_t)n * ld + k] : 0.f;
    out[i] = (__bf16)v;
}

#if HAVE_TDM
// Build a D# (ISA 08_async_tensor §8) for a rows x 32 bf16 tile whose rows are
// `strideElems` apart, and issue TENSOR_LOAD_TO_LDS.  Per-wave op, EXEC ignored.
// 6-arg builtin form: (g0, g1, g2, g3, g_ext, cpol).
__device__ __forceinline__
void tdm_load_tile_bf16(const __bf16* gptr, unsigned lds_off, int rows,
                        int strideElems)
{
    unsigned long long ga = (unsigned long long)(size_t)gptr;   // byte address
    u32x4 g0;
    g0[0] = 1u;                                   // count=1, user mode
    g0[1] = lds_off;                              // lds_addr (bytes)
    g0[2] = (unsigned)(ga & 0xffffffffu);         // global_addr[31:0]
    g0[3] = (unsigned)((ga >> 32) & 0x01ffffffu)  // global_addr[56:32]
          | (2u << 30);                           // type = 2 ("image")
    unsigned td0 = (unsigned)strideElems;         // tensor_dim0 (elements)
    unsigned td1 = 0x7fffffffu;                   // tensor_dim1: huge (no OOB)
    i32x8 g1;
    g1[0] = (int)(1u << 16);                      // wg_mask=0, data_size=1 (2B)
    g1[1] = (int)((td0 & 0xffffu) << 16);         // bits 63:48  = td0[15:0]
    g1[2] = (int)((td0 >> 16) & 0xffffu)          // bits 79:64  = td0[31:16]
          | (int)((td1 & 0xffffu) << 16);         // bits 95:80  = td1[15:0]
    g1[3] = (int)((td1 >> 16) & 0xffffu)          // bits 111:96 = td1[31:16]
          | (int)(BK << 16);                      // bits 127:112 tile_dim0 = 32
    g1[4] = rows & 0xffff;                        // tile_dim1 = rows, tile_dim2=0
    g1[5] = (int)td0;                             // tensor_dim0_stride[31:0]
    g1[6] = 0;                                    // stride[47:32]=0, dim1_stride lo
    g1[7] = 0;
    i32x4 z4 = {0, 0, 0, 0};
    i32x8 z8 = {0, 0, 0, 0, 0, 0, 0, 0};
    __builtin_amdgcn_tensor_load_to_lds(g0, g1, z4, z4, z8, 0);
}
#endif

// ------------------------------------------------------------------- GEMM ---
// C[M,Nreal](f32) = A[M,K](bf16) * W[Npad,K](bf16)^T + epilogue.
// 64x128 block tile, 8 waves: wave w -> rows 16*(w>>1), col tiles 4*(w&1)+{0..3}.
// TDM path: double-buffered LDS, DMA of chunk k+1 overlaps WMMA on chunk k.
__global__ __launch_bounds__(256)
void gemm_bf16_wmma(const __bf16* __restrict__ A, const __bf16* __restrict__ W,
                    float* __restrict__ C, __bf16* __restrict__ Cbf,
                    const float* __restrict__ bias, const float* __restrict__ resid,
                    int M, int K, int Nreal, int epi)
{
    __shared__ __bf16 sA[2][BM * BK];   // 2 x 4 KB
    __shared__ __bf16 sB[2][BN * BK];   // 2 x 8 KB

    const int m0   = blockIdx.y * BM;
    const int n0   = blockIdx.x * BN;
    const int tid  = threadIdx.x;
    const int lane = tid & 31;
    const int wave = tid >> 5;
    const int wr   = wave >> 1;      // row group 0..3
    const int wc   = wave & 1;       // col half 0..1
    const int r    = lane & 15;
    const int hi   = lane >> 4;

    v8f acc[4];
#pragma unroll
    for (int i = 0; i < 4; ++i) acc[i] = (v8f){};

    int cur = 0;
#if HAVE_TDM
    const unsigned sA_off[2] = { (unsigned)(size_t)&sA[0][0],
                                 (unsigned)(size_t)&sA[1][0] };
    const unsigned sB_off[2] = { (unsigned)(size_t)&sB[0][0],
                                 (unsigned)(size_t)&sB[1][0] };
    // Prologue: DMA chunk 0 into buffer 0.
    if (wave == 0) {
        tdm_load_tile_bf16(A + (size_t)m0 * K, sA_off[0], BM, K);
        tdm_load_tile_bf16(W + (size_t)n0 * K, sB_off[0], BN, K);
        __builtin_amdgcn_s_wait_tensorcnt(0);
    }
    __syncthreads();
#else
    const int lar = tid >> 2;          // A: row 0..63, one uint4 per thread
    const int laq = (tid & 3) * 8;
    const int lbr = tid >> 1;          // B: row 0..127, two uint4 per thread
    const int lbq = (tid & 1) * 16;
#endif

    for (int kb = 0; kb < K; kb += BK) {
#if HAVE_TDM
        const int nxt = cur ^ 1;
        if (wave == 0 && kb + BK < K) {
            // Issue next chunk's DMA; no wait -> overlaps with WMMA below.
            tdm_load_tile_bf16(A + (size_t)m0 * K + kb + BK, sA_off[nxt], BM, K);
            tdm_load_tile_bf16(W + (size_t)n0 * K + kb + BK, sB_off[nxt], BN, K);
        } else if (wave == 1 && kb + 2 * BK < K) {
            __builtin_prefetch(A + (size_t)(m0 + lane) * K + kb + 2 * BK, 0, 0);
            __builtin_prefetch(W + (size_t)(n0 + lane) * K + kb + 2 * BK, 0, 0);
        }
#else
        __syncthreads();
        *(uint4*)(&sA[0][lar * BK + laq]) =
            *(const uint4*)(A + (size_t)(m0 + lar) * K + kb + laq);
        *(uint4*)(&sB[0][lbr * BK + lbq]) =
            *(const uint4*)(W + (size_t)(n0 + lbr) * K + kb + lbq);
        *(uint4*)(&sB[0][lbr * BK + lbq + 8]) =
            *(const uint4*)(W + (size_t)(n0 + lbr) * K + kb + lbq + 8);
        if (kb + BK < K) {
            __builtin_prefetch(A + (size_t)(m0 + lar) * K + kb + BK + laq, 0, 0);
            __builtin_prefetch(W + (size_t)(n0 + lbr) * K + kb + BK + lbq, 0, 0);
        }
        __syncthreads();
#endif

        // ISA 16-bit A-matrix 16x32 layout: lane<16 -> K{0..7,16..23},
        // lane>=16 -> K{8..15,24..31}.
        const __bf16* pA = &sA[cur][0];
        const __bf16* pB = &sB[cur][0];
        v16bf af;
#pragma unroll
        for (int p = 0; p < 8; ++p) {
            int k = ((p & 4) ? 16 : 0) + (hi ? 8 : 0) + 2 * (p & 3);
            af[2 * p]     = pA[(16 * wr + r) * BK + k];
            af[2 * p + 1] = pA[(16 * wr + r) * BK + k + 1];
        }
#pragma unroll
        for (int ct = 0; ct < 4; ++ct) {
            v16bf bfg;
            int brow = 16 * (4 * wc + ct) + r;
#pragma unroll
            for (int p = 0; p < 8; ++p) {
                int k = ((p & 4) ? 16 : 0) + (hi ? 8 : 0) + 2 * (p & 3);
                bfg[2 * p]     = pB[brow * BK + k];
                bfg[2 * p + 1] = pB[brow * BK + k + 1];
            }
            acc[ct] = __builtin_amdgcn_wmma_f32_16x16x32_bf16(
                false, af, false, bfg, (short)0, acc[ct], false, false);
        }

#if HAVE_TDM
        __syncthreads();                       // all waves done with buf[cur]
        if (wave == 0 && kb + BK < K)
            __builtin_amdgcn_s_wait_tensorcnt(0);   // buf[nxt] DMA complete
        __syncthreads();                       // publish buf[nxt]
        cur = nxt;
#endif
    }

#pragma unroll
    for (int ct = 0; ct < 4; ++ct) {
        int col = n0 + 16 * (4 * wc + ct) + r;
        if (col >= Nreal) continue;
        float bv = 0.f;
        if (epi == EPI_BIAS_SOFTPLUS || epi == EPI_BIAS_GELU || epi == EPI_BIAS_RESID)
            bv = bias[col];
#pragma unroll
        for (int v = 0; v < 8; ++v) {
            int row = m0 + 16 * wr + hi * 8 + v;
            float cv = acc[ct][v] + bv;
            if (epi == EPI_BIAS_SOFTPLUS) {
                cv = (cv > 20.f) ? cv : log1pf(expf(cv));
            } else if (epi == EPI_BIAS_GELU) {
                float c3 = cv * cv * cv;
                cv = 0.5f * cv * (1.f + tanhf(0.7978845608028654f * (cv + 0.044715f * c3)));
            }
            if (epi == EPI_RESID || epi == EPI_BIAS_RESID)
                cv += resid[(size_t)row * Nreal + col];
            C[(size_t)row * Nreal + col] = cv;
            if (Cbf) Cbf[(size_t)row * Nreal + col] = (__bf16)cv;
        }
    }
}

// ------------------------------------------------------ depthwise conv4 -----
__global__ __launch_bounds__(256)
void conv4_silu(const float* __restrict__ xz, const float* __restrict__ cw,
                const float* __restrict__ cb, float* __restrict__ xc,
                __bf16* __restrict__ xc_bf)
{
    int c = blockIdx.x * 256 + threadIdx.x;   // 0 .. B*DINNER-1
    int b = c / DINNER;
    int d = c % DINNER;
    float w0 = cw[d * 4 + 0], w1 = cw[d * 4 + 1];
    float w2 = cw[d * 4 + 2], w3 = cw[d * 4 + 3];
    float bb = cb[d];
    float x0 = 0.f, x1 = 0.f, x2 = 0.f;
    for (int t = 0; t < SEQL; ++t) {
        size_t m = (size_t)b * SEQL + t;
        float x3 = xz[m * (2 * DINNER) + d];
        float s  = w0 * x0 + w1 * x1 + w2 * x2 + w3 * x3 + bb;
        s = s / (1.f + expf(-s));             // SiLU
        xc[m * DINNER + d]    = s;
        xc_bf[m * DINNER + d] = (__bf16)s;
        x0 = x1; x1 = x2; x2 = x3;
    }
}

// ------------------------------------------------- selective scan + gate ----
__global__ __launch_bounds__(256)
void scan_gate(const float* __restrict__ delta, const float* __restrict__ xc,
               const float* __restrict__ x_dbl, const float* __restrict__ xz,
               const float* __restrict__ A_log, const float* __restrict__ Dp_,
               __bf16* __restrict__ y_bf)
{
    __shared__ float sBC[32];
    int c = blockIdx.x * 256 + threadIdx.x;   // blocks never straddle batches
    int b = c / DINNER;
    int d = c % DINNER;
    float Ar[DSTATE];
#pragma unroll
    for (int n = 0; n < DSTATE; ++n) Ar[n] = -expf(A_log[d * DSTATE + n]);
    float Dp = Dp_[d];
    float h[DSTATE];
#pragma unroll
    for (int n = 0; n < DSTATE; ++n) h[n] = 0.f;

    for (int t = 0; t < SEQL; ++t) {
        size_t m = (size_t)b * SEQL + t;
        __syncthreads();
        if (threadIdx.x < 32)
            sBC[threadIdx.x] = x_dbl[m * 80 + DTRANK + threadIdx.x];
        __syncthreads();
        float dv = delta[m * DINNER + d];
        float xv = xc[m * DINNER + d];
        float a2 = 0.f;
#pragma unroll
        for (int n = 0; n < DSTATE; ++n) {
            float dA = expf(dv * Ar[n]);
            h[n] = dA * h[n] + dv * sBC[n] * xv;
            a2 += h[n] * sBC[DSTATE + n];
        }
        float y = a2 + xv * Dp;
        float z = xz[m * (2 * DINNER) + DINNER + d];
        y *= z / (1.f + expf(-z));
        y_bf[m * DINNER + d] = (__bf16)y;
    }
}

// ------------------------------------------------------------ layernorm -----
__global__ __launch_bounds__(256)
void layernorm768(const float* __restrict__ in, const float* __restrict__ w,
                  const float* __restrict__ b, float* __restrict__ out,
                  __bf16* __restrict__ outbf)
{
    __shared__ float red[16];
    int row = blockIdx.x;
    const float* x = in + (size_t)row * EMBED;
    float s = 0.f, s2 = 0.f;
    for (int i = threadIdx.x; i < EMBED; i += 256) {
        float v = x[i]; s += v; s2 += v * v;
    }
#pragma unroll
    for (int off = 16; off > 0; off >>= 1) {
        s  += __shfl_down(s,  off, 32);
        s2 += __shfl_down(s2, off, 32);
    }
    if ((threadIdx.x & 31) == 0) {
        red[threadIdx.x >> 5]       = s;
        red[8 + (threadIdx.x >> 5)] = s2;
    }
    __syncthreads();
    if (threadIdx.x == 0) {
        float a = 0.f, c2 = 0.f;
        for (int i = 0; i < 8; ++i) { a += red[i]; c2 += red[8 + i]; }
        float mu  = a * (1.f / EMBED);
        float var = c2 * (1.f / EMBED) - mu * mu;
        red[0] = mu; red[1] = rsqrtf(var + 1e-5f);
    }
    __syncthreads();
    float mu = red[0], rs = red[1];
    for (int i = threadIdx.x; i < EMBED; i += 256) {
        float v = (x[i] - mu) * rs * w[i] + b[i];
        out[(size_t)row * EMBED + i] = v;
        if (outbf) outbf[(size_t)row * EMBED + i] = (__bf16)v;
    }
}

// ------------------------------------------------------------------ host ----
extern "C" void kernel_launch(void* const* d_in, const int* in_sizes, int n_in,
                              void* d_out, int out_size, void* d_ws, size_t ws_size,
                              hipStream_t stream)
{
    const float* x         = (const float*)d_in[0];
    const float* in_proj_w = (const float*)d_in[1];
    const float* conv_w    = (const float*)d_in[2];
    const float* conv_b    = (const float*)d_in[3];
    const float* x_proj_w  = (const float*)d_in[4];
    const float* dt_proj_w = (const float*)d_in[5];
    const float* dt_proj_b = (const float*)d_in[6];
    const float* A_log     = (const float*)d_in[7];
    const float* D_param   = (const float*)d_in[8];
    const float* out_proj_w= (const float*)d_in[9];
    const float* ln1_w     = (const float*)d_in[10];
    const float* ln1_b     = (const float*)d_in[11];
    const float* ln2_w     = (const float*)d_in[12];
    const float* ln2_b     = (const float*)d_in[13];
    const float* ffn_w1    = (const float*)d_in[14];
    const float* ffn_b1    = (const float*)d_in[15];
    const float* ffn_w2    = (const float*)d_in[16];
    const float* ffn_b2    = (const float*)d_in[17];

    char* ws = (char*)d_ws;
    size_t off = 0;
    auto carve = [&](size_t bytes) -> void* {
        void* p = ws + off;
        off += (bytes + 255) & ~(size_t)255;
        return p;
    };

    float* xz    = (float*)carve((size_t)MROWS * 2 * DINNER * 4);
    float* xc    = (float*)carve((size_t)MROWS * DINNER * 4);
    float* x_dbl = (float*)carve((size_t)MROWS * 80 * 4);
    float* delta = (float*)carve((size_t)MROWS * DINNER * 4);
    float* hsum  = (float*)carve((size_t)MROWS * EMBED * 4);
    float* x1    = (float*)carve((size_t)MROWS * EMBED * 4);
    float* fbuf  = (float*)carve((size_t)MROWS * FFDIM * 4);
    float* gbuf  = (float*)carve((size_t)MROWS * EMBED * 4);
    __bf16* x_bf   = (__bf16*)carve((size_t)MROWS * EMBED * 2);
    __bf16* inw_bf = (__bf16*)carve((size_t)(2 * DINNER) * EMBED * 2);
    __bf16* xc_bf  = (__bf16*)carve((size_t)MROWS * DINNER * 2);
    __bf16* xpw_bf = (__bf16*)carve((size_t)128 * DINNER * 2);
    __bf16* dt_bf  = (__bf16*)carve((size_t)MROWS * 64 * 2);
    __bf16* dtw_bf = (__bf16*)carve((size_t)DINNER * 64 * 2);
    __bf16* y_bf   = (__bf16*)carve((size_t)MROWS * DINNER * 2);
    __bf16* ow_bf  = (__bf16*)carve((size_t)EMBED * DINNER * 2);
    __bf16* x1_bf  = (__bf16*)carve((size_t)MROWS * EMBED * 2);
    __bf16* w1_bf  = (__bf16*)carve((size_t)FFDIM * EMBED * 2);
    __bf16* f_bf   = (__bf16*)carve((size_t)MROWS * FFDIM * 2);
    __bf16* w2_bf  = (__bf16*)carve((size_t)EMBED * FFDIM * 2);

    auto cvt = [&](const float* in, __bf16* o, int N, int K, int ld, int Kp, int Npad) {
        long long total = (long long)Npad * Kp;
        cvt_pad_bf16<<<(unsigned)((total + 255) / 256), 256, 0, stream>>>(
            in, o, N, K, ld, Kp, total);
    };

    cvt(x,          x_bf,   MROWS,    EMBED,  EMBED,  EMBED,  MROWS);
    cvt(in_proj_w,  inw_bf, 2*DINNER, EMBED,  EMBED,  EMBED,  2*DINNER);
    cvt(x_proj_w,   xpw_bf, 80,       DINNER, DINNER, DINNER, 128);
    cvt(dt_proj_w,  dtw_bf, DINNER,   DTRANK, DTRANK, 64,     DINNER);
    cvt(out_proj_w, ow_bf,  EMBED,    DINNER, DINNER, DINNER, EMBED);
    cvt(ffn_w1,     w1_bf,  FFDIM,    EMBED,  EMBED,  EMBED,  FFDIM);
    cvt(ffn_w2,     w2_bf,  EMBED,    FFDIM,  FFDIM,  FFDIM,  EMBED);

    dim3 blk(256);
    // 1) in_proj: xz = x @ in_proj_w^T            [2048 x 3072]
    gemm_bf16_wmma<<<dim3(2 * DINNER / BN, MROWS / BM), blk, 0, stream>>>(
        x_bf, inw_bf, xz, nullptr, nullptr, nullptr, MROWS, EMBED, 2 * DINNER, EPI_NONE);
    // 2) causal conv4 + SiLU
    conv4_silu<<<dim3(NBATCH * DINNER / 256), blk, 0, stream>>>(
        xz, conv_w, conv_b, xc, xc_bf);
    // 3) x_proj: x_dbl = xc @ x_proj_w^T          [2048 x 80]
    gemm_bf16_wmma<<<dim3(128 / BN, MROWS / BM), blk, 0, stream>>>(
        xc_bf, xpw_bf, x_dbl, nullptr, nullptr, nullptr, MROWS, DINNER, 80, EPI_NONE);
    // 4) dt slice -> bf16, K padded 48 -> 64
    cvt(x_dbl, dt_bf, MROWS, DTRANK, 80, 64, MROWS);
    // 5) dt_proj + softplus -> delta              [2048 x 1536]
    gemm_bf16_wmma<<<dim3(DINNER / BN, MROWS / BM), blk, 0, stream>>>(
        dt_bf, dtw_bf, delta, nullptr, dt_proj_b, nullptr, MROWS, 64, DINNER, EPI_BIAS_SOFTPLUS);
    // 6) selective scan + D skip + SiLU(z) gate -> y_bf
    scan_gate<<<dim3(NBATCH * DINNER / 256), blk, 0, stream>>>(
        delta, xc, x_dbl, xz, A_log, D_param, y_bf);
    // 7) out_proj + residual x -> hsum            [2048 x 768]
    gemm_bf16_wmma<<<dim3(EMBED / BN, MROWS / BM), blk, 0, stream>>>(
        y_bf, ow_bf, hsum, nullptr, nullptr, x, MROWS, DINNER, EMBED, EPI_RESID);
    // 8) LN1 -> x1 (+bf16)
    layernorm768<<<dim3(MROWS), blk, 0, stream>>>(hsum, ln1_w, ln1_b, x1, x1_bf);
    // 9) FFN1 + bias + gelu(tanh) -> f (+bf16)    [2048 x 3072]
    gemm_bf16_wmma<<<dim3(FFDIM / BN, MROWS / BM), blk, 0, stream>>>(
        x1_bf, w1_bf, fbuf, f_bf, ffn_b1, nullptr, MROWS, EMBED, FFDIM, EPI_BIAS_GELU);
    // 10) FFN2 + bias + residual x1 -> g          [2048 x 768]
    gemm_bf16_wmma<<<dim3(EMBED / BN, MROWS / BM), blk, 0, stream>>>(
        f_bf, w2_bf, gbuf, nullptr, ffn_b2, x1, MROWS, FFDIM, EMBED, EPI_BIAS_RESID);
    // 11) LN2 -> d_out
    layernorm768<<<dim3(MROWS), blk, 0, stream>>>(gbuf, ln2_w, ln2_b, (float*)d_out, nullptr);
}
